// HybridLayer_48627619725837
// MI455X (gfx1250) — 16-bit and 8-bit matrix operand order repaired and checked by reading, compiled, not run on hardware
//
#include <hip/hip_runtime.h>
#include <cstdint>
#include <cstddef>

#define BDIM   2
#define NTOK   2048
#define DMODEL 1024
#define NHEAD  16
#define HEADD  64
#define NEXP   8
#define DFF    2048
#define BN     (BDIM * NTOK)
#define EPSF   1e-5f

typedef __attribute__((ext_vector_type(8)))  int    v8i;
typedef __attribute__((ext_vector_type(8)))  float  v8f;
typedef __attribute__((ext_vector_type(16))) __bf16 v16bf;

struct alignas(8)  U64x  { unsigned x, y; };
struct alignas(16) U128x { unsigned x, y, z, w; };
struct alignas(16) F4x   { float x, y, z, w; };
struct A8frag  { U64x a, b, c, d; };
struct B8frag  { U128x a, b; };
struct BFfrag  { U128x a, b; };

__device__ inline __bf16 f2bf(float f) { return (__bf16)f; }
__device__ inline unsigned short f2bfbits(float f) {
  return __builtin_bit_cast(unsigned short, (__bf16)f);
}
__device__ inline v8i ldA8(const int8_t* p) {
  A8frag f;
  f.a = *(const U64x*)(p);      f.b = *(const U64x*)(p + 8);
  f.c = *(const U64x*)(p + 32); f.d = *(const U64x*)(p + 40);
  return __builtin_bit_cast(v8i, f);
}
__device__ inline v8i ldB8(const int8_t* p) {
  B8frag f;
  f.a = *(const U128x*)(p); f.b = *(const U128x*)(p + 32);
  return __builtin_bit_cast(v8i, f);
}

__device__ inline float blockReduceSum(float v, float* sh) {
  int tid = threadIdx.x;
  sh[tid] = v; __syncthreads();
  for (int o = 128; o > 0; o >>= 1) { if (tid < o) sh[tid] += sh[tid + o]; __syncthreads(); }
  float r = sh[0]; __syncthreads();
  return r;
}
__device__ inline float blockReduceMax(float v, float* sh) {
  int tid = threadIdx.x;
  sh[tid] = v; __syncthreads();
  for (int o = 128; o > 0; o >>= 1) { if (tid < o) sh[tid] = fmaxf(sh[tid], sh[tid + o]); __syncthreads(); }
  float r = sh[0]; __syncthreads();
  return r;
}


__global__ void k_zero(float* p, int n) {
  int i = blockIdx.x * blockDim.x + threadIdx.x;
  if (i < n) p[i] = 0.f;
}

__global__ __launch_bounds__(256) void k_wabs(const float* __restrict__ w, long n, float* __restrict__ slot) {
  __shared__ float sh[256];
  float s = 0.f;
  for (long i = (long)blockIdx.x * blockDim.x + threadIdx.x; i < n; i += (long)gridDim.x * blockDim.x)
    s += fabsf(w[i]);
  s = blockReduceSum(s, sh);
  if (threadIdx.x == 0) atomicAdd(slot, s);
}

__global__ __launch_bounds__(256) void k_wquant(const float* __restrict__ w, long n,
                                                const float* __restrict__ slot, float invn,
                                                int8_t* __restrict__ q) {
  float mean = fmaxf(slot[0] * invn, 1e-5f);
  float scale = 1.f / mean;
  for (long i = (long)blockIdx.x * blockDim.x + threadIdx.x; i < n; i += (long)gridDim.x * blockDim.x) {
    float v = rintf(w[i] * scale);
    v = fminf(fmaxf(v, -1.f), 1.f);
    q[i] = (int8_t)v;
  }
}

__global__ __launch_bounds__(256) void k_ln_rows(const float* __restrict__ x,
                                                 const float* __restrict__ wt, const float* __restrict__ bs,
                                                 float* __restrict__ out, int C) {
  __shared__ float sh[256];
  const int row = blockIdx.x;
  const float* p = x + (size_t)row * C;
  float s = 0.f;
  for (int i = threadIdx.x; i < C; i += 256) s += p[i];
  s = blockReduceSum(s, sh);
  float mu = s / (float)C;
  float s2 = 0.f;
  for (int i = threadIdx.x; i < C; i += 256) { float d = p[i] - mu; s2 += d * d; }
  s2 = blockReduceSum(s2, sh);
  float rs = rsqrtf(s2 / (float)C + EPSF);
  float* o = out + (size_t)row * C;
  for (int i = threadIdx.x; i < C; i += 256)
    o[i] = (p[i] - mu) * rs * wt[i] + bs[i];
}

__global__ __launch_bounds__(256) void k_bstats(const float* __restrict__ x, long perBatch,
                                                float* __restrict__ stats) {
  __shared__ float sh[256];
  const int b = blockIdx.y;
  const float* p = x + (size_t)b * perBatch;
  float s1 = 0.f, s2 = 0.f;
  for (long i = (long)blockIdx.x * 256 + threadIdx.x; i < perBatch; i += (long)gridDim.x * 256) {
    float v = p[i]; s1 += v; s2 += v * v;
  }
  s1 = blockReduceSum(s1, sh);
  s2 = blockReduceSum(s2, sh);
  if (threadIdx.x == 0) { atomicAdd(&stats[2 * b], s1); atomicAdd(&stats[2 * b + 1], s2); }
}

template<int GLB>
__global__ __launch_bounds__(256) void k_rowquant(const float* __restrict__ x, int C,
                                                  const float* __restrict__ stats, int rowsPerBatch,
                                                  float invCount,
                                                  int8_t* __restrict__ q8, float* __restrict__ sAct,
                                                  const int* __restrict__ cntGuard) {
  if (cntGuard && (int)blockIdx.x >= cntGuard[0]) return;
  __shared__ float sh[256];
  const int row = blockIdx.x;
  const float* p = x + (size_t)row * C;
  float mu, rs;
  if constexpr (GLB) {
    int b = row / rowsPerBatch;
    float m1 = stats[2 * b] * invCount;
    float m2 = stats[2 * b + 1] * invCount;
    mu = m1;
    rs = rsqrtf(fmaxf(m2 - m1 * m1, 0.f) + EPSF);
  } else {
    float s = 0.f;
    for (int i = threadIdx.x; i < C; i += 256) s += p[i];
    s = blockReduceSum(s, sh);
    mu = s / (float)C;
    float s2 = 0.f;
    for (int i = threadIdx.x; i < C; i += 256) { float d = p[i] - mu; s2 += d * d; }
    s2 = blockReduceSum(s2, sh);
    rs = rsqrtf(s2 / (float)C + EPSF);
  }
  float am = 0.f;
  for (int i = threadIdx.x; i < C; i += 256) am = fmaxf(am, fabsf((p[i] - mu) * rs));
  am = blockReduceMax(am, sh);
  float s = 127.f / fmaxf(am, 1e-5f);
  int8_t* q = q8 + (size_t)row * C;
  for (int i = threadIdx.x; i < C; i += 256) {
    float qv = rintf((p[i] - mu) * rs * s);
    qv = fminf(fmaxf(qv, -128.f), 127.f);
    q[i] = (int8_t)qv;
  }
  if (threadIdx.x == 0) sAct[row] = s;
}

__global__ void k_mul(const float* __restrict__ a, const float* __restrict__ b,
                      float* __restrict__ o, long n) {
  for (long i = (long)blockIdx.x * blockDim.x + threadIdx.x; i < n; i += (long)gridDim.x * blockDim.x)
    o[i] = a[i] * b[i];
}

__global__ __launch_bounds__(256) void k_router(const float* __restrict__ h2,
                                                const float* __restrict__ rw,
                                                int* __restrict__ cnt,
                                                int* __restrict__ idx,
                                                float* __restrict__ cfl) {
  int t = blockIdx.x * blockDim.x + threadIdx.x;
  if (t >= BN) return;
  const float* xr = h2 + (size_t)t * DMODEL;
  float acc[NEXP];
#pragma unroll
  for (int e = 0; e < NEXP; ++e) acc[e] = 0.f;
  for (int i = 0; i < DMODEL; ++i) {
    float xv = xr[i];
#pragma unroll
    for (int e = 0; e < NEXP; ++e) acc[e] += xv * rw[e * DMODEL + i];
  }
  float mx = acc[0];
#pragma unroll
  for (int e = 1; e < NEXP; ++e) mx = fmaxf(mx, acc[e]);
  float p[NEXP], sum = 0.f;
#pragma unroll
  for (int e = 0; e < NEXP; ++e) { p[e] = expf(acc[e] - mx); sum += p[e]; }
  float inv = 1.f / sum;
#pragma unroll
  for (int e = 0; e < NEXP; ++e) p[e] *= inv;
  int i1 = 0;
#pragma unroll
  for (int e = 1; e < NEXP; ++e) if (p[e] > p[i1]) i1 = e;
  int i2 = (i1 == 0) ? 1 : 0;
#pragma unroll
  for (int e = 0; e < NEXP; ++e) if (e != i1 && p[e] > p[i2]) i2 = e;
  int p1 = atomicAdd(&cnt[i1], 1);
  idx[i1 * BN + p1] = t; cfl[i1 * BN + p1] = p[i1];
  int p2 = atomicAdd(&cnt[i2], 1);
  idx[i2 * BN + p2] = t; cfl[i2 * BN + p2] = p[i2];
}


enum { EPI_F32 = 0, EPI_SILU = 1, EPI_ADDRES = 2, EPI_MOE = 3, EPI_QKV = 4, EPI_VT = 5 };

template<int EPI, int GATHER>
__global__ __launch_bounds__(256) void k_gemm_q8(
    const int8_t* __restrict__ Aq, const float* __restrict__ sAct,
    const int8_t* __restrict__ Wq, const float* __restrict__ wsum, float winv,
    int N, int K,
    float* __restrict__ outF, const float* __restrict__ res,
    unsigned short* __restrict__ outBF,
    const int* __restrict__ gcnt, const int* __restrict__ gidx,
    const float* __restrict__ gcoef) {
  const int lane = threadIdx.x & 31;
  const int wave = threadIdx.x >> 5;
  const int wm = wave >> 2, wn = wave & 3;
  const int m0 = blockIdx.y * 64 + wm * 32;
  const int n0 = blockIdx.x * 128 + wn * 32;
  if (n0 >= N) return;
  int c = 0x7fffffff;
  if constexpr (GATHER || EPI == EPI_MOE) {
    c = gcnt[0];
    if (m0 >= c) return;
  }
  const float wfac = fmaxf(wsum[0] * winv, 1e-5f);
  const int l15 = lane & 15;
  const int hi = (lane >> 4) & 1;

  int tok0 = m0 + l15, tok1 = m0 + 16 + l15;
  if constexpr (GATHER) {
    tok0 = (m0 + l15 < c) ? gidx[m0 + l15] : 0;
    tok1 = (m0 + 16 + l15 < c) ? gidx[m0 + 16 + l15] : 0;
  }
  const int8_t* ap0 = Aq + (size_t)tok0 * K + hi * 16;
  const int8_t* ap1 = Aq + (size_t)tok1 * K + hi * 16;
  const int8_t* wp0 = Wq + (size_t)(n0 + l15) * K + hi * 16;
  const int8_t* wp1 = wp0 + (size_t)16 * K;

  v8i acc[2][2] = {};
  for (int k0 = 0; k0 < K; k0 += 64) {
    v8i a0 = ldA8(ap0), a1 = ldA8(ap1);
    v8i b0 = ldB8(wp0), b1 = ldB8(wp1);
    __builtin_prefetch(ap0 + 256, 0, 1);
    __builtin_prefetch(ap1 + 256, 0, 1);
    acc[0][0] = __builtin_amdgcn_wmma_i32_16x16x64_iu8(true, a0, true, b0, acc[0][0], false, false);
    acc[0][1] = __builtin_amdgcn_wmma_i32_16x16x64_iu8(true, a0, true, b1, acc[0][1], false, false);
    acc[1][0] = __builtin_amdgcn_wmma_i32_16x16x64_iu8(true, a1, true, b0, acc[1][0], false, false);
    acc[1][1] = __builtin_amdgcn_wmma_i32_16x16x64_iu8(true, a1, true, b1, acc[1][1], false, false);
    ap0 += 64; ap1 += 64; wp0 += 64; wp1 += 64;
  }
#pragma unroll
  for (int mt = 0; mt < 2; ++mt)
#pragma unroll
    for (int nt = 0; nt < 2; ++nt)
#pragma unroll
      for (int r = 0; r < 8; ++r) {
        const int row = m0 + mt * 16 + r + hi * 8;
        const int col = n0 + nt * 16 + l15;
        int srow = row;
        if constexpr (GATHER) srow = (row < c) ? gidx[row] : 0;
        float val = (float)acc[mt][nt][r] * (wfac / sAct[srow]);
        if constexpr (EPI == EPI_F32) {
          outF[(size_t)row * N + col] = val;
        } else if constexpr (EPI == EPI_SILU) {
          float e = __builtin_amdgcn_exp2f(-1.44269504f * val);
          outF[(size_t)row * N + col] = val / (1.f + e);
        } else if constexpr (EPI == EPI_ADDRES) {
          size_t ix = (size_t)row * N + col;
          outF[ix] = res[ix] + val;
        } else if constexpr (EPI == EPI_MOE) {
          if (row < c) {
            int token = gidx[row];
            outF[(size_t)token * N + col] += gcoef[row] * val;
          }
        } else if constexpr (EPI == EPI_QKV) {
          int b = row / NTOK, n = row % NTOK;
          int h = col >> 6, hd = col & 63;
          outBF[((((size_t)b * NHEAD + h) * NTOK + n) << 6) + hd] = f2bfbits(val);
        } else {
          int b = row / NTOK, n = row % NTOK;
          int h = col >> 6, hd = col & 63;
          outBF[(((size_t)b * NHEAD + h) * HEADD + hd) * NTOK + n] = f2bfbits(val);
        }
      }
}


__global__ __launch_bounds__(128) void k_attn(const unsigned short* __restrict__ qb,
                                              const unsigned short* __restrict__ kb,
                                              const unsigned short* __restrict__ vT,
                                              float* __restrict__ y) {
  __shared__ float sls[4][16][32];
  const int lane = threadIdx.x & 31;
  const int wave = threadIdx.x >> 5;
  const int l15 = lane & 15;
  const int hi = (lane >> 4) & 1;
  const int bh = blockIdx.x;
  const int b = bh / NHEAD, h = bh % NHEAD;
  const int n0 = blockIdx.y * 64 + wave * 16;

  const float l32 = -3.46573590280f, l512 = -6.23832462504f;
  const float gamma = 1.f - expf(l32 + (l512 - l32) * (float)h / (float)(NHEAD - 1));
  const float log2g = log2f(gamma);
  float gpow[8];
  gpow[0] = 0.125f;
#pragma unroll
  for (int r = 1; r < 8; ++r) gpow[r] = gpow[r - 1] * gamma;

  const unsigned short* qrow = qb + ((size_t)bh * NTOK + n0 + l15) * HEADD + hi * 8;
  BFfrag qf0, qf1;
  qf0.a = *(const U128x*)(qrow);      qf0.b = *(const U128x*)(qrow + 16);
  qf1.a = *(const U128x*)(qrow + 32); qf1.b = *(const U128x*)(qrow + 48);
  v16bf aq0 = __builtin_bit_cast(v16bf, qf0);
  v16bf aq1 = __builtin_bit_cast(v16bf, qf1);

  v8f yac[4] = {};
  for (int m0 = 0; m0 < n0 + 16; m0 += 32) {
#pragma unroll
    for (int sub = 0; sub < 2; ++sub) {
      const int mc = m0 + sub * 16;
      v8f s = {};
      const unsigned short* krow = kb + ((size_t)bh * NTOK + mc + l15) * HEADD + hi * 8;
      BFfrag kf0, kf1;
      kf0.a = *(const U128x*)(krow);      kf0.b = *(const U128x*)(krow + 16);
      kf1.a = *(const U128x*)(krow + 32); kf1.b = *(const U128x*)(krow + 48);
      s = __builtin_amdgcn_wmma_f32_16x16x32_bf16(false, aq0, false,
            __builtin_bit_cast(v16bf, kf0), (short)0, s, false, false);
      s = __builtin_amdgcn_wmma_f32_16x16x32_bf16(false, aq1, false,
            __builtin_bit_cast(v16bf, kf1), (short)0, s, false, false);
      const int d0 = n0 + hi * 8 - mc - l15;
      const float G0 = __builtin_amdgcn_exp2f((float)d0 * log2g);
      if (mc + 15 <= n0) {
#pragma unroll
        for (int r = 0; r < 8; ++r)
          sls[wave][r + hi * 8][sub * 16 + l15] = s[r] * (G0 * gpow[r]);
      } else {
#pragma unroll
        for (int r = 0; r < 8; ++r) {
          float dec = (d0 + r < 0) ? 0.f : (G0 * gpow[r]);
          sls[wave][r + hi * 8][sub * 16 + l15] = s[r] * dec;
        }
      }
    }
    F4x u0 = *(const F4x*)&sls[wave][l15][hi * 8];
    F4x u1 = *(const F4x*)&sls[wave][l15][hi * 8 + 4];
    F4x u2 = *(const F4x*)&sls[wave][l15][16 + hi * 8];
    F4x u3 = *(const F4x*)&sls[wave][l15][16 + hi * 8 + 4];
    float f[16] = {u0.x, u0.y, u0.z, u0.w, u1.x, u1.y, u1.z, u1.w,
                   u2.x, u2.y, u2.z, u2.w, u3.x, u3.y, u3.z, u3.w};
    v16bf sA;
#pragma unroll
    for (int j = 0; j < 16; ++j) sA[j] = f2bf(f[j]);
#pragma unroll
    for (int t = 0; t < 4; ++t) {
      const unsigned short* vrow = vT + ((size_t)bh * HEADD + t * 16 + l15) * NTOK + m0 + hi * 8;
      BFfrag vf;
      vf.a = *(const U128x*)(vrow); vf.b = *(const U128x*)(vrow + 16);
      yac[t] = __builtin_amdgcn_wmma_f32_16x16x32_bf16(false, sA, false,
                 __builtin_bit_cast(v16bf, vf), (short)0, yac[t], false, false);
    }
  }
#pragma unroll
  for (int t = 0; t < 4; ++t)
#pragma unroll
    for (int r = 0; r < 8; ++r) {
      int n = n0 + r + hi * 8;
      int col = h * HEADD + t * 16 + l15;
      y[((size_t)b * NTOK + n) * DMODEL + col] = yac[t][r];
    }
}


extern "C" void kernel_launch(void* const* d_in, const int* in_sizes, int n_in,
                              void* d_out, int out_size, void* d_ws, size_t ws_size,
                              hipStream_t stream) {
  const float* x      = (const float*)d_in[0];
  const float* q_w    = (const float*)d_in[1];
  const float* k_w    = (const float*)d_in[2];
  const float* v_w    = (const float*)d_in[3];
  const float* g_w    = (const float*)d_in[4];
  const float* o_w    = (const float*)d_in[5];
  const float* ret_nw = (const float*)d_in[6];
  const float* ret_nb = (const float*)d_in[7];
  const float* n1_w   = (const float*)d_in[8];
  const float* n1_b   = (const float*)d_in[9];
  const float* n2_w   = (const float*)d_in[10];
  const float* n2_b   = (const float*)d_in[11];
  const float* rt_w   = (const float*)d_in[12];
  const float* exp_w1 = (const float*)d_in[13];
  const float* exp_w2 = (const float*)d_in[14];
  float* out = (float*)d_out;

  char* base = (char*)d_ws;
  size_t off = 0;
  auto alloc = [&](size_t bytes) -> char* {
    char* p = base + off;
    off = (off + bytes + 255) & ~(size_t)255;
    return p;
  };

  const size_t dd  = (size_t)DMODEL * DMODEL;
  const size_t fdd = (size_t)DFF * DMODEL;

  float*  stats = (float*)alloc(1024);
  int*    ecnt  = (int*)(stats + 24);
  float*  hbuf  = (float*)alloc((size_t)BN * DMODEL * 4);
  int8_t* q8    = (int8_t*)alloc((size_t)BN * DMODEL);
  float*  sA    = (float*)alloc((size_t)BN * 4);
  float*  sT    = (float*)alloc((size_t)BN * 4);
  int8_t* wq5   = (int8_t*)alloc(5 * dd);
  int8_t* w1q   = (int8_t*)alloc(NEXP * fdd);
  int8_t* w2q   = (int8_t*)alloc(NEXP * fdd);
  unsigned short* qbf = (unsigned short*)alloc((size_t)BN * DMODEL * 2);
  unsigned short* kbf = (unsigned short*)alloc((size_t)BN * DMODEL * 2);
  unsigned short* vtb = (unsigned short*)alloc((size_t)BN * DMODEL * 2);
  float*  gbuf  = (float*)alloc((size_t)BN * DMODEL * 4);
  float*  ybuf  = (float*)alloc((size_t)BN * DMODEL * 4);
  float*  ylnb  = (float*)alloc((size_t)BN * DMODEL * 4);
  float*  x1b   = (float*)alloc((size_t)BN * DMODEL * 4);
  float*  h2b   = (float*)alloc((size_t)BN * DMODEL * 4);
  int*    eidx  = (int*)alloc((size_t)NEXP * BN * 4);
  float*  ecoef = (float*)alloc((size_t)NEXP * BN * 4);
  float*  t1    = (float*)alloc((size_t)BN * DFF * 4);
  int8_t* t1q8  = (int8_t*)alloc((size_t)BN * DFF);

  k_zero<<<1, 256, 0, stream>>>(stats, 256);

  const float* projW[5] = {q_w, k_w, v_w, g_w, o_w};
  for (int i = 0; i < 5; ++i)
    k_wabs<<<64, 256, 0, stream>>>(projW[i], (long)dd, stats + i);
  for (int e = 0; e < NEXP; ++e) {
    k_wabs<<<128, 256, 0, stream>>>(exp_w1 + (size_t)e * fdd, (long)fdd, stats + 5 + e);
    k_wabs<<<128, 256, 0, stream>>>(exp_w2 + (size_t)e * fdd, (long)fdd, stats + 13 + e);
  }
  for (int i = 0; i < 5; ++i)
    k_wquant<<<256, 256, 0, stream>>>(projW[i], (long)dd, stats + i, 1.f / (float)dd, wq5 + i * dd);
  for (int e = 0; e < NEXP; ++e) {
    k_wquant<<<256, 256, 0, stream>>>(exp_w1 + (size_t)e * fdd, (long)fdd, stats + 5 + e, 1.f / (float)fdd, w1q + e * fdd);
    k_wquant<<<256, 256, 0, stream>>>(exp_w2 + (size_t)e * fdd, (long)fdd, stats + 13 + e, 1.f / (float)fdd, w2q + e * fdd);
  }

  k_ln_rows<<<BN, 256, 0, stream>>>(x, n1_w, n1_b, hbuf, DMODEL);
  k_bstats<<<dim3(64, BDIM), 256, 0, stream>>>(hbuf, (long)NTOK * DMODEL, stats + 32);
  const float invND = 1.f / ((float)NTOK * (float)DMODEL);
  k_rowquant<1><<<BN, 256, 0, stream>>>(hbuf, DMODEL, stats + 32, NTOK, invND, q8, sA, nullptr);

  dim3 g1(DMODEL / 128, BN / 64);
  const float invDD = 1.f / (float)dd;
  k_gemm_q8<EPI_QKV, 0><<<g1, 256, 0, stream>>>(q8, sA, wq5 + 0 * dd, stats + 0, invDD, DMODEL, DMODEL, nullptr, nullptr, qbf, nullptr, nullptr, nullptr);
  k_gemm_q8<EPI_QKV, 0><<<g1, 256, 0, stream>>>(q8, sA, wq5 + 1 * dd, stats + 1, invDD, DMODEL, DMODEL, nullptr, nullptr, kbf, nullptr, nullptr, nullptr);
  k_gemm_q8<EPI_VT , 0><<<g1, 256, 0, stream>>>(q8, sA, wq5 + 2 * dd, stats + 2, invDD, DMODEL, DMODEL, nullptr, nullptr, vtb, nullptr, nullptr, nullptr);
  k_gemm_q8<EPI_SILU,0><<<g1, 256, 0, stream>>>(q8, sA, wq5 + 3 * dd, stats + 3, invDD, DMODEL, DMODEL, gbuf, nullptr, nullptr, nullptr, nullptr, nullptr);

  k_attn<<<dim3(BDIM * NHEAD, NTOK / 64), 128, 0, stream>>>(qbf, kbf, vtb, ybuf);

  k_ln_rows<<<BN, 256, 0, stream>>>(ybuf, ret_nw, ret_nb, ylnb, DMODEL);
  k_mul<<<512, 256, 0, stream>>>(ylnb, gbuf, gbuf, (long)BN * DMODEL);
  k_bstats<<<dim3(64, BDIM), 256, 0, stream>>>(gbuf, (long)NTOK * DMODEL, stats + 36);
  k_rowquant<1><<<BN, 256, 0, stream>>>(gbuf, DMODEL, stats + 36, NTOK, invND, q8, sA, nullptr);
  k_gemm_q8<EPI_ADDRES, 0><<<g1, 256, 0, stream>>>(q8, sA, wq5 + 4 * dd, stats + 4, invDD, DMODEL, DMODEL, x1b, x, nullptr, nullptr, nullptr, nullptr);

  k_ln_rows<<<BN, 256, 0, stream>>>(x1b, n2_w, n2_b, h2b, DMODEL);
  k_router<<<BN / 256, 256, 0, stream>>>(h2b, rt_w, ecnt, eidx, ecoef);
  k_rowquant<0><<<BN, 256, 0, stream>>>(h2b, DMODEL, nullptr, 1, 0.f, q8, sA, nullptr);

  hipMemcpyAsync(out, x1b, (size_t)BN * DMODEL * 4, hipMemcpyDeviceToDevice, stream);

  dim3 ge1(DFF / 128, BN / 64), ge2(DMODEL / 128, BN / 64);
  const float invFD = 1.f / (float)fdd;
  for (int e = 0; e < NEXP; ++e) {
    k_gemm_q8<EPI_SILU, 1><<<ge1, 256, 0, stream>>>(q8, sA, w1q + e * fdd, stats + 5 + e, invFD, DFF, DMODEL,
                                                    t1, nullptr, nullptr, ecnt + e, eidx + e * BN, nullptr);
    k_rowquant<0><<<BN, 256, 0, stream>>>(t1, DFF, nullptr, 1, 0.f, t1q8, sT, ecnt + e);
    k_gemm_q8<EPI_MOE, 0><<<ge2, 256, 0, stream>>>(t1q8, sT, w2q + e * fdd, stats + 13 + e, invFD, DMODEL, DFF,
                                                   out, nullptr, nullptr, ecnt + e, eidx + e * BN, ecoef + e * BN);
  }
}
